// SpatialFeatureEncoder_41128606826990
// MI455X (gfx1250) — matrix operand order repaired and checked by reading, not compiled, not run on hardware
//
#include <hip/hip_runtime.h>
#include <hip/hip_bf16.h>

typedef __attribute__((ext_vector_type(16))) _Float16 v16h;
typedef __attribute__((ext_vector_type(8)))  _Float16 v8h;
typedef __attribute__((ext_vector_type(8)))  float    v8f;

// ---------------------------------------------------------------------------
// Model constants
// ---------------------------------------------------------------------------
#define BB   16      // batch
#define NP   2048    // dense points
#define DD   768     // model dim
#define HH   12      // heads
#define KQN  256     // fps samples / visual tokens
#define LTN  32      // text tokens
#define JTN  288     // joint tokens (256 + 32)

// ---------------------------------------------------------------------------
// CDNA5 async global->LDS copy (ASYNCcnt-tracked). LDS operand is the low
// 32 bits of the flat pointer (flat->LDS mapping truncates to addr[31:0]).
// ---------------------------------------------------------------------------
__device__ __forceinline__ void async_copy16(float* lds, const float* g)
{
    unsigned l32 = (unsigned)(unsigned long long)lds;
    asm volatile("global_load_async_to_lds_b128 %0, %1, off"
                 :: "v"(l32), "v"(g) : "memory");
}
__device__ __forceinline__ void async_wait_all()
{
    asm volatile("s_wait_asynccnt 0x0" ::: "memory");
}

// ---------------------------------------------------------------------------
// WMMA GEMM:  C[M,N] = act( A[M,K] @ W[N,K]^T + bias [+ addC] )
// BM=128, BN=128, BK=32; 256 threads = 8 waves (2 M-rows x 4 N-cols),
// wave tile 64x32 = 4x2 wmma frags = 8 v_wmma per K-step.
// Assumes M%128==0, N%128==0, K%32==0 (true for every call site).
// ---------------------------------------------------------------------------
#define GBM 128
#define GBN 128
#define GBK 32
#define ASTR 40    // halfs per A row in LDS (padded)
#define BSTR 136   // halfs per B row in LDS (padded), B stored [K][N]

__global__ __launch_bounds__(256) void wmma_gemm(
    const float* __restrict__ A, int lda,
    const float* __restrict__ W, int ldw,
    const float* __restrict__ bias,
    const float* __restrict__ addC,
    float* __restrict__ C, int ldc,
    int M, int N, int K, int relu)
{
    __shared__ __align__(16) _Float16 As[GBM * ASTR];
    __shared__ __align__(16) _Float16 Bs[GBK * BSTR];

    const int tid  = threadIdx.x;
    const int wid  = tid >> 5;
    const int lane = tid & 31;
    const int r    = lane & 15;   // frag row (A) / frag col (B,D) / k sub
    const int kh   = lane >> 4;   // lane half

    const long m0 = (long)blockIdx.y * GBM;
    const long n0 = (long)blockIdx.x * GBN;
    const int  wm = (wid >> 2) * 64;  // wave M offset (0 or 64)
    const int  wn = (wid & 3) * 32;   // wave N offset (0,32,64,96)

    v8f zf = {0.f,0.f,0.f,0.f,0.f,0.f,0.f,0.f};
    v8f acc[4][2];
    #pragma unroll
    for (int i = 0; i < 4; i++) { acc[i][0] = zf; acc[i][1] = zf; }

    // A loader: thread -> (row, 16-col half)
    const int arow = tid >> 1;           // 0..127
    const int acol = (tid & 1) * 16;     // 0 or 16
    const float* aptr = A + (m0 + arow) * (long)lda + acol;
    // B loader: thread -> (n pair, 8-k chunk); stores transposed + packed
    const int bpair = tid & 63;          // n = 2*bpair, 2*bpair+1
    const int bkq   = (tid >> 6) * 8;    // k offset 0,8,16,24
    const float* bptr0 = W + (n0 + 2 * bpair) * (long)ldw + bkq;
    const float* bptr1 = bptr0 + ldw;

    for (int k0 = 0; k0 < K; k0 += GBK) {
        __syncthreads();
        // ---- stage A tile (row-major, f32 -> f16) ----
        {
            const float4* p = (const float4*)(aptr + k0);
            float4 f0 = p[0], f1 = p[1], f2 = p[2], f3 = p[3];
            if (k0 + GBK < K) __builtin_prefetch(aptr + k0 + GBK, 0, 1);
            v8h h0, h1;
            h0[0]=(_Float16)f0.x; h0[1]=(_Float16)f0.y; h0[2]=(_Float16)f0.z; h0[3]=(_Float16)f0.w;
            h0[4]=(_Float16)f1.x; h0[5]=(_Float16)f1.y; h0[6]=(_Float16)f1.z; h0[7]=(_Float16)f1.w;
            h1[0]=(_Float16)f2.x; h1[1]=(_Float16)f2.y; h1[2]=(_Float16)f2.z; h1[3]=(_Float16)f2.w;
            h1[4]=(_Float16)f3.x; h1[5]=(_Float16)f3.y; h1[6]=(_Float16)f3.z; h1[7]=(_Float16)f3.w;
            *(v8h*)&As[arow * ASTR + acol]     = h0;
            *(v8h*)&As[arow * ASTR + acol + 8] = h1;
        }
        // ---- stage B tile transposed/packed: Bs[k][n] = W[n0+n][k0+k] ----
        {
            const float4* p0 = (const float4*)(bptr0 + k0);
            const float4* p1 = (const float4*)(bptr1 + k0);
            float4 a0 = p0[0], a1 = p0[1];
            float4 c0 = p1[0], c1 = p1[1];
            float r0[8] = {a0.x,a0.y,a0.z,a0.w, a1.x,a1.y,a1.z,a1.w};
            float r1[8] = {c0.x,c0.y,c0.z,c0.w, c1.x,c1.y,c1.z,c1.w};
            #pragma unroll
            for (int e = 0; e < 8; e++) {
                union { _Float16 h[2]; unsigned u; } pk;
                pk.h[0] = (_Float16)r0[e];
                pk.h[1] = (_Float16)r1[e];
                *(unsigned*)&Bs[(bkq + e) * BSTR + 2 * bpair] = pk.u;
            }
        }
        __syncthreads();

        // ---- build fragments (ISA 7.12.2 layouts) ----
        v16h aF[4], bF[2];
        #pragma unroll
        for (int i = 0; i < 4; i++) {
            const int row = wm + i * 16 + r;
            v8h lo = *(const v8h*)&As[row * ASTR + kh * 8];
            v8h hi = *(const v8h*)&As[row * ASTR + 16 + kh * 8];
            aF[i] = __builtin_shufflevector(lo, hi, 0,1,2,3,4,5,6,7,8,9,10,11,12,13,14,15);
        }
        #pragma unroll
        for (int j = 0; j < 2; j++) {
            const int nc = wn + j * 16 + r;
            #pragma unroll
            for (int s = 0; s < 8; s++) {
                bF[j][s]     = Bs[(kh * 8 + s) * BSTR + nc];
                bF[j][8 + s] = Bs[(16 + kh * 8 + s) * BSTR + nc];
            }
        }
        // ---- 4x2 WMMA ----
        #pragma unroll
        for (int i = 0; i < 4; i++)
            #pragma unroll
            for (int j = 0; j < 2; j++)
                acc[i][j] = __builtin_amdgcn_wmma_f32_16x16x32_f16(
                    false, aF[i], false, bF[j], (short)0, acc[i][j], false, false);
    }

    // ---- epilogue: bias + residual + relu, C/D layout from ISA ----
    #pragma unroll
    for (int i = 0; i < 4; i++) {
        #pragma unroll
        for (int j = 0; j < 2; j++) {
            const long col = n0 + wn + j * 16 + r;
            const float bv = bias ? bias[col] : 0.f;
            #pragma unroll
            for (int e = 0; e < 8; e++) {
                const long row = m0 + wm + i * 16 + kh * 8 + e;
                float v = acc[i][j][e] + bv;
                if (addC) v += addC[row * (long)ldc + col];
                if (relu) v = v > 0.f ? v : 0.f;
                C[row * (long)ldc + col] = v;
            }
        }
    }
}

// ---------------------------------------------------------------------------
// Fused attention (flash-style, online softmax). One block per (b,h),
// one thread per query row. K/V staged to LDS with async-tensor path.
// ---------------------------------------------------------------------------
__global__ __launch_bounds__(JTN) void attn_kernel(
    const float* __restrict__ Qb, long qB, int qLd,
    const float* __restrict__ Kb, long kB, int kLd,
    const float* __restrict__ Vb, long vB, int vLd,
    const unsigned char* __restrict__ mask, int visCount, int Lt,
    float* __restrict__ Ob, long oB, int oLd,
    int H, int Lq, int Lk)
{
    __shared__ __align__(16) float Ks[32 * 64];
    __shared__ __align__(16) float Vs[32 * 64];
    __shared__ float Ss[JTN * 32];

    const int z = blockIdx.x;
    const int b = z / H, h = z % H;
    const int t = threadIdx.x;          // query row (blockDim == Lq)
    const int nthr = blockDim.x;

    const float* qp = Qb + (long)b * qB + (long)t * qLd + h * 64;
    float qr[64];
    #pragma unroll
    for (int d = 0; d < 64; d += 4) {
        float4 f = *(const float4*)(qp + d);
        qr[d] = f.x; qr[d+1] = f.y; qr[d+2] = f.z; qr[d+3] = f.w;
    }
    float o[64];
    #pragma unroll
    for (int d = 0; d < 64; d++) o[d] = 0.f;
    float m = -3.0e38f, l = 0.f;
    float* srow = &Ss[t * 32];
    const float scale = 0.125f;   // 1/sqrt(64)

    for (int j0 = 0; j0 < Lk; j0 += 32) {
        __syncthreads();   // previous tile fully consumed before overwrite
        for (int i = t; i < 32 * 16; i += nthr) {
            const int jj = i >> 4;
            const int d4 = (i & 15) * 4;
            const long j = j0 + jj;
            async_copy16(&Ks[jj * 64 + d4], Kb + (long)b * kB + j * kLd + h * 64 + d4);
            async_copy16(&Vs[jj * 64 + d4], Vb + (long)b * vB + j * vLd + h * 64 + d4);
        }
        async_wait_all();
        __syncthreads();

        float mx = -3.0e38f;
        for (int jj = 0; jj < 32; jj++) {
            float s = 0.f;
            const float* kp = &Ks[jj * 64];
            #pragma unroll
            for (int d = 0; d < 64; d++) s += qr[d] * kp[d];
            s *= scale;
            const int j = j0 + jj;
            if (mask && j >= visCount && mask[b * Lt + (j - visCount)] == 0) s = -1.0e30f;
            srow[jj] = s;
            mx = fmaxf(mx, s);
        }
        const float mnew = fmaxf(m, mx);
        const float corr = __expf(m - mnew);
        l *= corr;
        #pragma unroll
        for (int d = 0; d < 64; d++) o[d] *= corr;
        for (int jj = 0; jj < 32; jj++) {
            const float p = __expf(srow[jj] - mnew);
            l += p;
            const float* vp = &Vs[jj * 64];
            #pragma unroll
            for (int d = 0; d < 64; d++) o[d] += p * vp[d];
        }
        m = mnew;
    }
    const float invl = 1.f / l;
    float* op = Ob + (long)b * oB + (long)t * oLd + h * 64;
    #pragma unroll
    for (int d = 0; d < 64; d++) op[d] = o[d] * invl;
}

// ---------------------------------------------------------------------------
// Farthest point sampling: one block per batch, LDS-resident points/dists.
// ---------------------------------------------------------------------------
__global__ __launch_bounds__(256) void fps_kernel(const float* __restrict__ xyz,
                                                  int* __restrict__ idx)
{
    __shared__ float xs[NP * 3];
    __shared__ float dist[NP];
    __shared__ float rv[256];
    __shared__ int   ri[256];
    const int b = blockIdx.x, tid = threadIdx.x;
    const float* base = xyz + (long)b * NP * 3;
    for (int i = tid; i < NP * 3; i += 256) xs[i] = base[i];
    for (int i = tid; i < NP; i += 256) dist[i] = 1e10f;
    __syncthreads();
    int cur = 0;
    if (tid == 0) idx[b * KQN] = 0;
    for (int s = 1; s < KQN; s++) {
        const float cx = xs[cur*3], cy = xs[cur*3+1], cz = xs[cur*3+2];
        float bv = -1.f; int bi = 0;
        for (int i = tid; i < NP; i += 256) {
            const float dx = xs[i*3]-cx, dy = xs[i*3+1]-cy, dz = xs[i*3+2]-cz;
            const float d2 = dx*dx + dy*dy + dz*dz;
            const float d = fminf(dist[i], d2);
            dist[i] = d;
            if (d > bv) { bv = d; bi = i; }
        }
        rv[tid] = bv; ri[tid] = bi;
        __syncthreads();
        for (int off = 128; off > 0; off >>= 1) {
            if (tid < off) {
                const float v2 = rv[tid+off]; const int i2 = ri[tid+off];
                if (v2 > rv[tid] || (v2 == rv[tid] && i2 < ri[tid])) { rv[tid]=v2; ri[tid]=i2; }
            }
            __syncthreads();
        }
        cur = ri[0];
        if (tid == 0) idx[b * KQN + s] = cur;
        __syncthreads();
    }
}

// ---------------------------------------------------------------------------
// Small elementwise / utility kernels
// ---------------------------------------------------------------------------
__global__ void gather_feat_kernel(const float* __restrict__ Z, const int* __restrict__ idx,
                                   float* __restrict__ pf)
{
    const long i = (long)blockIdx.x * blockDim.x + threadIdx.x;
    if (i >= (long)BB * KQN * DD) return;
    const int d = (int)(i % DD);
    const int k = (int)((i / DD) % KQN);
    const int b = (int)(i / ((long)DD * KQN));
    pf[i] = Z[((long)b * NP + idx[b * KQN + k]) * DD + d];
}

__global__ void gather_pos_kernel(const float* __restrict__ xyz, const int* __restrict__ idx,
                                  float* __restrict__ pp, float* __restrict__ out_pp)
{
    const int i = blockIdx.x * blockDim.x + threadIdx.x;
    if (i >= BB * KQN * 3) return;
    const int d = i % 3;
    const int k = (i / 3) % KQN;
    const int b = i / (3 * KQN);
    const float v = xyz[((long)b * NP + idx[b * KQN + k]) * 3 + d];
    pp[i] = v;
    out_pp[i] = v;
}

__global__ void posh_kernel(const float* __restrict__ xyz, const float* __restrict__ w1,
                            const float* __restrict__ b1, const float* __restrict__ bng,
                            const float* __restrict__ bnb, float* __restrict__ out, long M)
{
    const long i = (long)blockIdx.x * blockDim.x + threadIdx.x;
    if (i >= M * DD) return;
    const long mr = i / DD;
    const int n = (int)(i % DD);
    const float x0 = xyz[mr*3], x1 = xyz[mr*3+1], x2 = xyz[mr*3+2];
    float h = x0 * w1[n*3] + x1 * w1[n*3+1] + x2 * w1[n*3+2] + b1[n];
    h = h * rsqrtf(1.f + 1e-5f) * bng[n] + bnb[n];
    out[i] = fmaxf(h, 0.f);
}

// LayerNorm over D=768; one block per row; optional residual input.
__global__ __launch_bounds__(256) void ln_kernel(
    const float* __restrict__ x, const float* __restrict__ res,
    const float* __restrict__ g, const float* __restrict__ bta,
    float* __restrict__ y)
{
    __shared__ float sb[256], qb[256];
    const long row = blockIdx.x;
    const int tid = threadIdx.x;
    const float* xr = x + row * DD;
    float v0 = xr[tid], v1 = xr[tid + 256], v2 = xr[tid + 512];
    if (res) {
        const float* rr = res + row * DD;
        v0 += rr[tid]; v1 += rr[tid + 256]; v2 += rr[tid + 512];
    }
    sb[tid] = v0 + v1 + v2;
    qb[tid] = v0*v0 + v1*v1 + v2*v2;
    __syncthreads();
    for (int off = 128; off > 0; off >>= 1) {
        if (tid < off) { sb[tid] += sb[tid+off]; qb[tid] += qb[tid+off]; }
        __syncthreads();
    }
    const float mean = sb[0] * (1.f / DD);
    const float var  = qb[0] * (1.f / DD) - mean * mean;
    const float inv  = rsqrtf(var + 1e-5f);
    float* yr = y + row * DD;
    yr[tid]       = (v0 - mean) * inv * g[tid]       + bta[tid];
    yr[tid + 256] = (v1 - mean) * inv * g[tid + 256] + bta[tid + 256];
    yr[tid + 512] = (v2 - mean) * inv * g[tid + 512] + bta[tid + 512];
}

// Batched strided copy: dst[c*dstStride + r] = src[c*srcStride + r]
__global__ void copy2d_kernel(float* __restrict__ dst, long dstStride,
                              const float* __restrict__ src, long srcStride,
                              long chunkFloats, int nChunks)
{
    const long total = chunkFloats * nChunks;
    const long i = (long)blockIdx.x * blockDim.x + threadIdx.x;
    if (i >= total) return;
    const long c = i / chunkFloats;
    const long r = i % chunkFloats;
    dst[c * dstStride + r] = src[c * srcStride + r];
}

__global__ void mean_kernel(const float* __restrict__ vis, float* __restrict__ vg)
{
    const int i = blockIdx.x * blockDim.x + threadIdx.x;
    if (i >= BB * DD) return;
    const int d = i % DD, b = i / DD;
    float s = 0.f;
    for (int k = 0; k < KQN; k++) s += vis[((long)b * KQN + k) * DD + d];
    vg[i] = s * (1.f / KQN);
}

__global__ __launch_bounds__(256) void tctx_kernel(const float* __restrict__ tg,
                                                   const float* __restrict__ txt,
                                                   float* __restrict__ tc)
{
    __shared__ float lg[LTN];
    const int b = blockIdx.x, tid = threadIdx.x;
    if (tid < LTN) {
        float s = 0.f;
        const float* tr = txt + ((long)b * LTN + tid) * DD;
        const float* gr = tg + (long)b * DD;
        for (int d = 0; d < DD; d++) s += gr[d] * tr[d];
        lg[tid] = s;
    }
    __syncthreads();
    if (tid == 0) {
        float mx = -3.0e38f;
        for (int l = 0; l < LTN; l++) mx = fmaxf(mx, lg[l]);
        float sum = 0.f;
        for (int l = 0; l < LTN; l++) { const float e = __expf(lg[l] - mx); lg[l] = e; sum += e; }
        const float inv = 1.f / sum;
        for (int l = 0; l < LTN; l++) lg[l] *= inv;
    }
    __syncthreads();
    for (int d = tid; d < DD; d += 256) {
        float s = 0.f;
        for (int l = 0; l < LTN; l++) s += lg[l] * txt[((long)b * LTN + l) * DD + d];
        tc[(long)b * DD + d] = s;
    }
}

__global__ __launch_bounds__(256) void pooler_kernel(
    const float* __restrict__ vg, const float* __restrict__ tc,
    const float* __restrict__ w1, const float* __restrict__ b1,
    const float* __restrict__ lng, const float* __restrict__ lnb,
    const float* __restrict__ w2, const float* __restrict__ b2,
    float* __restrict__ out)
{
    __shared__ float xin[2 * DD];
    __shared__ float hb[DD];
    __shared__ float red[256], red2[256];
    const int b = blockIdx.x, tid = threadIdx.x;
    for (int i = tid; i < DD; i += 256) {
        xin[i] = vg[(long)b * DD + i];
        xin[DD + i] = tc[(long)b * DD + i];
    }
    __syncthreads();
    for (int n = tid; n < DD; n += 256) {
        float s = b1[n];
        const float* wr = w1 + (long)n * (2 * DD);
        for (int k = 0; k < 2 * DD; k++) s += xin[k] * wr[k];
        hb[n] = s;
    }
    __syncthreads();
    float s = 0.f, ss = 0.f;
    for (int i = tid; i < DD; i += 256) { const float v = hb[i]; s += v; ss += v * v; }
    red[tid] = s; red2[tid] = ss;
    __syncthreads();
    for (int off = 128; off > 0; off >>= 1) {
        if (tid < off) { red[tid] += red[tid+off]; red2[tid] += red2[tid+off]; }
        __syncthreads();
    }
    const float mean = red[0] * (1.f / DD);
    const float var  = red2[0] * (1.f / DD) - mean * mean;
    const float inv  = rsqrtf(var + 1e-5f);
    for (int i = tid; i < DD; i += 256)
        hb[i] = fmaxf((hb[i] - mean) * inv * lng[i] + lnb[i], 0.f);
    __syncthreads();
    for (int n = tid; n < DD; n += 256) {
        float s2 = b2[n];
        const float* wr = w2 + (long)n * DD;
        for (int k = 0; k < DD; k++) s2 += hb[k] * wr[k];
        out[(long)b * DD + n] = s2;
    }
}

// ---------------------------------------------------------------------------
// Host-side orchestration
// ---------------------------------------------------------------------------
static inline void gemm(hipStream_t s, const float* A, int lda, const float* W, int ldw,
                        const float* bias, const float* addC, float* C, int ldc,
                        int M, int N, int K, int relu)
{
    dim3 g(N / GBN, M / GBM);
    wmma_gemm<<<g, 256, 0, s>>>(A, lda, W, ldw, bias, addC, C, ldc, M, N, K, relu);
}

static inline int blks(long n) { return (int)((n + 255) / 256); }

extern "C" void kernel_launch(void* const* d_in, const int* in_sizes, int n_in,
                              void* d_out, int out_size, void* d_ws, size_t ws_size,
                              hipStream_t stream)
{
    (void)in_sizes; (void)n_in; (void)out_size; (void)ws_size;

    // ---- inputs (setup_inputs order) ----
    const float* Z     = (const float*)d_in[0];   // [16,2048,768]
    const float* xyz   = (const float*)d_in[1];   // [16,2048,3]
    const float* textf = (const float*)d_in[2];   // [16,32,768]
    const float* tg    = (const float*)d_in[3];   // [16,768]
    const unsigned char* tmask = (const unsigned char*)d_in[4]; // [16,32] bool

    // ---- params: jax pytree flatten, dict keys sorted alphabetically ----
    const float* fm_b    = (const float*)d_in[5];
    const float* fm_w    = (const float*)d_in[6];
    const float* fpos_b1 = (const float*)d_in[7];
    const float* fpos_b2 = (const float*)d_in[8];
    const float* fpos_bnb= (const float*)d_in[9];
    const float* fpos_bng= (const float*)d_in[10];
    const float* fpos_w1 = (const float*)d_in[11];
    const float* fpos_w2 = (const float*)d_in[12];
    const float* fpre_b  = (const float*)d_in[13];
    const float* fpre_g  = (const float*)d_in[14];
    auto LP = [&](int l, int j) { return (const float*)d_in[15 + 18 * l + j]; };
    const float* map_b   = (const float*)d_in[87];
    const float* map_w   = (const float*)d_in[88];
    const float* pool_b1 = (const float*)d_in[89];
    const float* pool_b2 = (const float*)d_in[90];
    const float* pool_lnb= (const float*)d_in[91];
    const float* pool_lng= (const float*)d_in[92];
    const float* pool_w1 = (const float*)d_in[93];
    const float* pool_w2 = (const float*)d_in[94];
    const float* pos_b1  = (const float*)d_in[95];
    const float* pos_b2  = (const float*)d_in[96];
    const float* pos_bnb = (const float*)d_in[97];
    const float* pos_bng = (const float*)d_in[98];
    const float* pos_w1  = (const float*)d_in[99];
    const float* pos_w2  = (const float*)d_in[100];
    const float* pre_b   = (const float*)d_in[101];
    const float* pre_g   = (const float*)d_in[102];

    // ---- workspace layout (floats) ----
    float* ws = (float*)d_ws;
    const long MFULL = (long)BB * NP;          // 32768 dense rows
    const long MVIS  = (long)BB * KQN;         // 4096 visual rows
    const long MJNT  = (long)BB * JTN;         // 4608 joint rows
    const size_t OFF_A     = 0;                                 // hid+posOut, later CA KV
    const size_t OFF_FULL  = OFF_A     + (size_t)MFULL * 1536;
    const size_t OFF_VIS   = OFF_FULL  + (size_t)MFULL * DD;
    const size_t OFF_TXT   = OFF_VIS   + (size_t)MVIS * DD;
    const size_t OFF_JOINT = OFF_TXT   + (size_t)BB * LTN * DD;
    const size_t OFF_QKV   = OFF_JOINT + (size_t)MJNT * DD;
    const size_t OFF_CTX   = OFF_QKV   + (size_t)MJNT * 2304;
    const size_t OFF_TMP   = OFF_CTX   + (size_t)MJNT * DD;
    const size_t OFF_FFH   = OFF_TMP   + (size_t)MJNT * DD;     // also aliases point_feats
    const size_t OFF_QCA   = OFF_FFH   + (size_t)MJNT * 3072;
    const size_t OFF_PP    = OFF_QCA   + (size_t)MVIS * DD;
    const size_t OFF_IDX   = OFF_PP    + (size_t)BB * KQN * 3;
    const size_t OFF_VG    = OFF_IDX   + (size_t)BB * KQN;
    const size_t OFF_TC    = OFF_VG    + (size_t)BB * DD;

    float* hid    = ws + OFF_A;
    float* posOut = ws + OFF_A + (size_t)MFULL * DD;
    float* kv     = ws + OFF_A;          // reused after preprocessing
    float* full   = ws + OFF_FULL;
    float* vis    = ws + OFF_VIS;
    float* txt    = ws + OFF_TXT;
    float* joint  = ws + OFF_JOINT;
    float* qkv    = ws + OFF_QKV;
    float* ctx    = ws + OFF_CTX;
    float* tmp    = ws + OFF_TMP;
    float* ffh    = ws + OFF_FFH;
    float* pf     = ws + OFF_FFH;        // alias (disjoint in time)
    float* qca    = ws + OFF_QCA;
    float* pp     = ws + OFF_PP;
    int*   idx    = (int*)(ws + OFF_IDX);
    float* vg     = ws + OFF_VG;
    float* tc     = ws + OFF_TC;

    float* out      = (float*)d_out;
    float* out_vis  = out;                               // [16,256,768]
    float* out_txt  = out + (size_t)MVIS * DD;           // [16,32,768]
    float* out_pool = out_txt + (size_t)BB * LTN * DD;   // [16,768]
    float* out_pp   = out_pool + (size_t)BB * DD;        // [16,256,3]

    // ================= preprocessing =================
    fps_kernel<<<BB, 256, 0, stream>>>(xyz, idx);
    gather_feat_kernel<<<blks((long)MVIS * DD), 256, 0, stream>>>(Z, idx, pf);
    gather_pos_kernel<<<blks((long)BB * KQN * 3), 256, 0, stream>>>(xyz, idx, pp, out_pp);

    // full = LN(Z @ full_map + pos_embed(xyz))
    posh_kernel<<<blks(MFULL * DD), 256, 0, stream>>>(xyz, fpos_w1, fpos_b1, fpos_bng, fpos_bnb, hid, MFULL);
    gemm(stream, hid, DD, fpos_w2, DD, fpos_b2, nullptr, posOut, DD, (int)MFULL, DD, DD, 0);
    gemm(stream, Z, DD, fm_w, DD, fm_b, posOut, full, DD, (int)MFULL, DD, DD, 0);
    ln_kernel<<<(int)MFULL, 256, 0, stream>>>(full, nullptr, fpre_g, fpre_b, full);

    // sp = LN(point_feats @ map + pos_embed(point_pos)) -> vis
    posh_kernel<<<blks(MVIS * DD), 256, 0, stream>>>(pp, pos_w1, pos_b1, pos_bng, pos_bnb, hid, MVIS);
    gemm(stream, hid, DD, pos_w2, DD, pos_b2, nullptr, posOut, DD, (int)MVIS, DD, DD, 0);
    gemm(stream, pf, DD, map_w, DD, map_b, posOut, tmp, DD, (int)MVIS, DD, DD, 0);
    ln_kernel<<<(int)MVIS, 256, 0, stream>>>(tmp, nullptr, pre_g, pre_b, vis);

    // txt = text_feats
    copy2d_kernel<<<blks((long)BB * LTN * DD), 256, 0, stream>>>(
        txt, (long)LTN * DD, textf, (long)LTN * DD, (long)LTN * DD, BB);

    // ================= transformer layers =================
    for (int l = 0; l < 4; l++) {
        const float* ca_in_b  = LP(l, 0);
        const float* ca_in_w  = LP(l, 1);
        const float* ca_out_b = LP(l, 2);
        const float* ca_out_w = LP(l, 3);
        const float* caln_b   = LP(l, 4);
        const float* caln_g   = LP(l, 5);
        const float* ffn_b1   = LP(l, 6);
        const float* ffn_b2   = LP(l, 7);
        const float* ffn_w1   = LP(l, 8);
        const float* ffn_w2   = LP(l, 9);
        const float* ffnln_b  = LP(l, 10);
        const float* ffnln_g  = LP(l, 11);
        const float* sa_in_b  = LP(l, 12);
        const float* sa_in_w  = LP(l, 13);
        const float* sa_out_b = LP(l, 14);
        const float* sa_out_w = LP(l, 15);
        const float* saln_b   = LP(l, 16);
        const float* saln_g   = LP(l, 17);

        // ---- cross attention: vis queries over dense `full` ----
        gemm(stream, full, DD, ca_in_w + (size_t)DD * DD, DD, ca_in_b + DD, nullptr,
             kv, 1536, (int)MFULL, 1536, DD, 0);                   // K|V packed [32768,1536]
        gemm(stream, vis, DD, ca_in_w, DD, ca_in_b, nullptr, qca, DD, (int)MVIS, DD, DD, 0);
        attn_kernel<<<BB * HH, KQN, 0, stream>>>(
            qca, (long)KQN * DD, DD,
            kv, (long)NP * 1536, 1536,
            kv + DD, (long)NP * 1536, 1536,
            nullptr, 0, 0,
            ctx, (long)KQN * DD, DD, HH, KQN, NP);
        gemm(stream, ctx, DD, ca_out_w, DD, ca_out_b, vis, tmp, DD, (int)MVIS, DD, DD, 0);
        ln_kernel<<<(int)MVIS, 256, 0, stream>>>(tmp, nullptr, caln_g, caln_b, vis);

        // ---- joint = concat(vis, txt) ----
        copy2d_kernel<<<blks((long)MVIS * DD), 256, 0, stream>>>(
            joint, (long)JTN * DD, vis, (long)KQN * DD, (long)KQN * DD, BB);
        copy2d_kernel<<<blks((long)BB * LTN * DD), 256, 0, stream>>>(
            joint + (size_t)KQN * DD, (long)JTN * DD, txt, (long)LTN * DD, (long)LTN * DD, BB);

        // ---- self attention over joint (with text key-padding mask) ----
        gemm(stream, joint, DD, sa_in_w, DD, sa_in_b, nullptr, qkv, 2304, (int)MJNT, 2304, DD, 0);
        attn_kernel<<<BB * HH, JTN, 0, stream>>>(
            qkv, (long)JTN * 2304, 2304,
            qkv + DD, (long)JTN * 2304, 2304,
            qkv + 2 * DD, (long)JTN * 2304, 2304,
            tmask, KQN, LTN,
            ctx, (long)JTN * DD, DD, HH, JTN, JTN);
        gemm(stream, ctx, DD, sa_out_w, DD, sa_out_b, joint, tmp, DD, (int)MJNT, DD, DD, 0);
        ln_kernel<<<(int)MJNT, 256, 0, stream>>>(tmp, nullptr, saln_g, saln_b, joint);

        // ---- FFN ----
        gemm(stream, joint, DD, ffn_w1, DD, ffn_b1, nullptr, ffh, 3072, (int)MJNT, 3072, DD, 1);
        gemm(stream, ffh, 3072, ffn_w2, 3072, ffn_b2, joint, tmp, DD, (int)MJNT, DD, 3072, 0);
        ln_kernel<<<(int)MJNT, 256, 0, stream>>>(tmp, nullptr, ffnln_g, ffnln_b, joint);

        // ---- split joint back into vis / txt ----
        copy2d_kernel<<<blks((long)MVIS * DD), 256, 0, stream>>>(
            vis, (long)KQN * DD, joint, (long)JTN * DD, (long)KQN * DD, BB);
        copy2d_kernel<<<blks((long)BB * LTN * DD), 256, 0, stream>>>(
            txt, (long)LTN * DD, joint + (size_t)KQN * DD, (long)JTN * DD, (long)LTN * DD, BB);
    }

    // ================= outputs =================
    copy2d_kernel<<<blks((long)MVIS * DD), 256, 0, stream>>>(
        out_vis, (long)MVIS * DD, vis, (long)MVIS * DD, (long)MVIS * DD, 1);
    copy2d_kernel<<<blks((long)BB * LTN * DD), 256, 0, stream>>>(
        out_txt, (long)BB * LTN * DD, txt, (long)BB * LTN * DD, (long)BB * LTN * DD, 1);

    mean_kernel<<<blks((long)BB * DD), 256, 0, stream>>>(vis, vg);
    tctx_kernel<<<BB, 256, 0, stream>>>(tg, txt, tc);
    pooler_kernel<<<BB, 256, 0, stream>>>(vg, tc, pool_w1, pool_b1, pool_lng, pool_lnb,
                                          pool_w2, pool_b2, out_pool);
}
